// SegmentLinear_3504693313636
// MI455X (gfx1250) — matrix operand order repaired and checked by reading, not compiled, not run on hardware
//
#include <hip/hip_runtime.h>

typedef __attribute__((ext_vector_type(16))) _Float16 v16h;
typedef __attribute__((ext_vector_type(4)))  _Float16 v4h;
typedef __attribute__((ext_vector_type(8)))  float    v8f;

#define TOK_N   65536
#define D_IN    512
#define D_OUT   512
#define SEG_N   16
#define TILE_M  16
#define MAX_TILES (TOK_N / TILE_M + SEG_N)   // 4112: sum of per-segment ceil-tiles upper bound

// ---------------- workspace layout ----------------
// [0, 8 MB)                : Wh  — f16 copy of W  (SEG_N * D_OUT * D_IN halves)
// [8 MB, +96 ints)         : meta — counts[0..15] | offsets[16..31] | cursors[32..47] | tileOff[48..64]
// [8 MB + 512 B, +N ints)  : idx — token ids, compacted per segment

// f32 -> f16 copy of W (one pass per launch; removes cvt from GEMM inner loop)
__global__ void seglin_cvtw(const float4* __restrict__ W4, v4h* __restrict__ Wh4, int n4) {
    int i = blockIdx.x * 256 + threadIdx.x;
    if (i >= n4) return;
    float4 v = W4[i];
    v4h h;
    h[0] = (_Float16)v.x; h[1] = (_Float16)v.y;
    h[2] = (_Float16)v.z; h[3] = (_Float16)v.w;
    Wh4[i] = h;
}

__global__ void seglin_zero(int* __restrict__ meta) {
    if (threadIdx.x < 96) meta[threadIdx.x] = 0;
}

__global__ void seglin_count(const int* __restrict__ coords, int* __restrict__ meta) {
    int n = blockIdx.x * 256 + threadIdx.x;
    atomicAdd(&meta[coords[n] & (SEG_N - 1)], 1);
}

__global__ void seglin_prefix(int* __restrict__ meta) {
    if (threadIdx.x == 0 && blockIdx.x == 0) {
        int off = 0, toff = 0;
        meta[48] = 0;
        for (int i = 0; i < SEG_N; ++i) {
            meta[16 + i] = off;
            off  += meta[i];
            toff += (meta[i] + TILE_M - 1) >> 4;
            meta[48 + i + 1] = toff;
        }
    }
}

__global__ void seglin_scatter(const int* __restrict__ coords, int* __restrict__ meta,
                               int* __restrict__ idx) {
    int n = blockIdx.x * 256 + threadIdx.x;
    int s = coords[n] & (SEG_N - 1);
    int pos = atomicAdd(&meta[32 + s], 1);
    idx[meta[16 + s] + pos] = n;
}

// Grouped GEMM tile kernel: one block = one 16-token tile of one segment,
// 8 waves x (16 tokens x 64 cols) accumulated via v_wmma_f32_16x16x32_f16.
__global__ __launch_bounds__(256) void seglin_gemm(
    const float*    __restrict__ x,
    const _Float16* __restrict__ Wh,
    const float*    __restrict__ bias,
    const int*      __restrict__ idx,
    const int*      __restrict__ meta,
    float*          __restrict__ out)
{
    __shared__ _Float16 Ash[TILE_M * D_IN];   // 16 KB: gathered A tile, f16
    __shared__ int toks[TILE_M];

    const int g   = blockIdx.x;
    const int tid = threadIdx.x;

    // locate (segment, tile-in-segment) from tile-offset prefix table
    int s = -1;
#pragma unroll
    for (int i = 0; i < SEG_N; ++i)
        if (g >= meta[48 + i] && g < meta[48 + i + 1]) s = i;
    if (s < 0) return;                         // uniform: whole block exits, EXEC stays full otherwise

    const int segStart = meta[16 + s];
    const int segCount = meta[s];
    const int rowBase  = segStart + (g - meta[48 + s]) * TILE_M;

    if (tid < TILE_M) {
        const int r = rowBase + tid;
        toks[tid] = (r < segStart + segCount) ? idx[r] : -1;
    }
    __syncthreads();

    // Stage A tile (16 x 512) into LDS as f16; coalesced float4 global reads.
#pragma unroll
    for (int j = 0; j < 8; ++j) {
        const int lin = tid + j * 256;         // float4 slot index, 2048 total
        const int row = lin >> 7;              // 128 float4 per row
        const int c4  = lin & 127;
        const int tok = toks[row];
        float4 v = make_float4(0.f, 0.f, 0.f, 0.f);
        if (tok >= 0) v = *(const float4*)(x + (size_t)tok * D_IN + (c4 << 2));
        v4h h;
        h[0] = (_Float16)v.x; h[1] = (_Float16)v.y;
        h[2] = (_Float16)v.z; h[3] = (_Float16)v.w;
        *(v4h*)(&Ash[row * D_IN + (c4 << 2)]) = h;
    }
    __syncthreads();

    const int lane  = tid & 31;
    const int laneN = lane & 15;
    const int hi    = lane >> 4;               // 0: lanes 0-15, 1: lanes 16-31
    const int colBase = (tid >> 5) * 64;       // 8 waves x 64 columns = 512

    const _Float16* __restrict__ Wseg = Wh + (size_t)s * D_OUT * D_IN;
    // Per-lane B row pointers: column fixed per lane, K contiguous in memory.
    const uint4* wq0 = (const uint4*)(Wseg + (size_t)(colBase +  0 + laneN) * D_IN + hi * 8);
    const uint4* wq1 = (const uint4*)(Wseg + (size_t)(colBase + 16 + laneN) * D_IN + hi * 8);
    const uint4* wq2 = (const uint4*)(Wseg + (size_t)(colBase + 32 + laneN) * D_IN + hi * 8);
    const uint4* wq3 = (const uint4*)(Wseg + (size_t)(colBase + 48 + laneN) * D_IN + hi * 8);
    const int aoff = laneN * D_IN + hi * 8;

    v8f acc0 = {}, acc1 = {}, acc2 = {}, acc3 = {};

    for (int k0 = 0; k0 < D_IN; k0 += 32) {
        union { v16h v; uint4 q[2]; } a, b0, b1, b2, b3;
        // A frag (ISA 16-bit A 16x32): halves 0..7 = K k0+hi*8+{0..7}, halves 8..15 = +16
        a.q[0] = *(const uint4*)(&Ash[aoff + k0]);
        a.q[1] = *(const uint4*)(&Ash[aoff + k0 + 16]);
        const int qi = k0 >> 3;                // uint4 index along K (8 halves each)
        b0.q[0] = wq0[qi]; b0.q[1] = wq0[qi + 2];
        b1.q[0] = wq1[qi]; b1.q[1] = wq1[qi + 2];
        b2.q[0] = wq2[qi]; b2.q[1] = wq2[qi + 2];
        b3.q[0] = wq3[qi]; b3.q[1] = wq3[qi + 2];

        acc0 = __builtin_amdgcn_wmma_f32_16x16x32_f16(false, a.v, false, b0.v, (short)0, acc0, false, false);
        acc1 = __builtin_amdgcn_wmma_f32_16x16x32_f16(false, a.v, false, b1.v, (short)0, acc1, false, false);
        acc2 = __builtin_amdgcn_wmma_f32_16x16x32_f16(false, a.v, false, b2.v, (short)0, acc2, false, false);
        acc3 = __builtin_amdgcn_wmma_f32_16x16x32_f16(false, a.v, false, b3.v, (short)0, acc3, false, false);
    }

    // C/D layout: VGPR r -> token row (r + 8*hi), column = colBase + t*16 + laneN
    const int mBase = hi * 8;
#pragma unroll
    for (int t = 0; t < 4; ++t) {
        const int col = colBase + t * 16 + laneN;
        const float bv = bias[s * D_OUT + col];
        const v8f av = (t == 0) ? acc0 : (t == 1) ? acc1 : (t == 2) ? acc2 : acc3;
#pragma unroll
        for (int r = 0; r < 8; ++r) {
            const int tok = toks[mBase + r];
            if (tok >= 0) out[(size_t)tok * D_OUT + col] = av[r] + bv;
        }
    }
}

extern "C" void kernel_launch(void* const* d_in, const int* in_sizes, int n_in,
                              void* d_out, int out_size, void* d_ws, size_t ws_size,
                              hipStream_t stream) {
    const float* x      = (const float*)d_in[0];
    const int*   coords = (const int*)d_in[1];
    const float* W      = (const float*)d_in[2];
    const float* bias   = (const float*)d_in[3];
    float* out = (float*)d_out;

    _Float16* Wh = (_Float16*)d_ws;
    int* meta = (int*)((char*)d_ws + (size_t)SEG_N * D_OUT * D_IN * sizeof(_Float16));
    int* idx  = meta + 128;

    const int wElems = SEG_N * D_OUT * D_IN;
    seglin_cvtw   <<<(wElems / 4 + 255) / 256, 256, 0, stream>>>((const float4*)W, (v4h*)Wh, wElems / 4);
    seglin_zero   <<<1, 128, 0, stream>>>(meta);
    seglin_count  <<<TOK_N / 256, 256, 0, stream>>>(coords, meta);
    seglin_prefix <<<1, 32, 0, stream>>>(meta);
    seglin_scatter<<<TOK_N / 256, 256, 0, stream>>>(coords, meta, idx);
    seglin_gemm   <<<MAX_TILES, 256, 0, stream>>>(x, Wh, bias, idx, meta, out);
}
